// self_Attention_2010044695017
// MI455X (gfx1250) — matrix operand order repaired and checked by reading, not compiled, not run on hardware
//
#include <hip/hip_runtime.h>

typedef __attribute__((ext_vector_type(16))) _Float16 v16h;
typedef __attribute__((ext_vector_type(8)))  float    v8f;
typedef __attribute__((ext_vector_type(4)))  unsigned int u32x4;
typedef __attribute__((ext_vector_type(8)))  int i32x8;
typedef __attribute__((ext_vector_type(4)))  int i32x4;

#define DEV static __device__ __forceinline__

constexpr int B_ = 2, S_ = 2048, E_ = 1024, H_ = 16, D_ = 64;
constexpr int BH_   = B_ * H_;           // 32 head-slices
constexpr int ROWS_ = B_ * H_ * S_;      // 65536 rows of D=64 (== B*S*E/64)
constexpr float SCALE_ = 0.03125f;       // 1/sqrt(EMBED)
constexpr int KP_ = 72;                  // LDS tile pitch in halves (TDM pad: +4 dwords / 32 dwords)

// workspace layout (in _Float16 elements)
constexpr size_t WS_Q  = 0;
constexpr size_t WS_K  = WS_Q  + (size_t)ROWS_ * D_;
constexpr size_t WS_VT = WS_K  + (size_t)ROWS_ * D_;   // V transposed per head: [BH][D][S]
constexpr size_t WS_O  = WS_VT + (size_t)ROWS_ * D_;   // attention out, head-flat == (B,S,E)
constexpr size_t WS_WO = WS_O  + (size_t)ROWS_ * D_;   // Wo in f16

DEV v8f wmma_f16(v16h a, v16h b, v8f c) {
  return __builtin_amdgcn_wmma_f32_16x16x32_f16(
      /*neg_a=*/false, a, /*neg_b=*/false, b,
      /*c_mod=*/(short)0, c, /*reuse_a=*/false, /*reuse_b=*/false);
}

// A fragment (16x32 f16): row = lane&15, K = g*16+i
DEV v16h load_a_f32(const float* base, int ld, int row, int k0, int g) {
  v16h a;
  const float* p = base + (size_t)row * ld + k0 + g * 16;
#pragma unroll
  for (int i = 0; i < 16; ++i) a[i] = (_Float16)p[i];
  return a;
}
DEV v16h load_a_f16(const _Float16* base, int ld, int row, int k0, int g) {
  v16h a;
  const _Float16* p = base + (size_t)row * ld + k0 + g * 16;
#pragma unroll
  for (int i = 0; i < 16; ++i) a[i] = p[i];
  return a;
}
// B fragment (32x16 f16) from a k-contiguous source: col = lane&15, K = g*16+i
DEV v16h load_bt_f32(const float* p) {
  v16h b;
#pragma unroll
  for (int i = 0; i < 16; ++i) b[i] = (_Float16)p[i];
  return b;
}
DEV v16h load_bt_f16(const _Float16* p) {
  v16h b;
#pragma unroll
  for (int i = 0; i < 16; ++i) b[i] = p[i];
  return b;
}
// C-fragment store through distinct registers (avoids per-store xcnt stalls)
DEV void store_c_f16(_Float16* dst, size_t stride, v8f acc) {
  _Float16 h[8];
#pragma unroll
  for (int r = 0; r < 8; ++r) h[r] = (_Float16)acc[r];
#pragma unroll
  for (int r = 0; r < 8; ++r) dst[(size_t)r * stride] = h[r];
}

// ---- TDM: 2-D tile load Global->LDS, 2-byte elements, padded rows (pitch 72 halves)
// D# per CDNA5 ISA ch.8: group0 {count=1, lds_addr, global_addr, type=2},
// group1 {data_size=1(2B), pad_enable, pad_interval=4(32dw), pad_amount=3(4dw),
//         tensor_dim0/1, tile_dim0/1, tensor_dim0_stride}
DEV void tdm_load_2d(const _Float16* gsrc, unsigned lds_off,
                     unsigned tensor_d0, unsigned tensor_d1,
                     unsigned long long stride0,
                     unsigned tile_d0, unsigned tile_d1) {
  unsigned long long ga = (unsigned long long)(uintptr_t)gsrc;
  u32x4 g0;
  g0[0] = 1u;                                              // count=1, no gather
  g0[1] = lds_off;                                         // lds_addr
  g0[2] = (unsigned)(ga & 0xffffffffu);                    // global_addr[31:0]
  g0[3] = (unsigned)((ga >> 32) & 0x01ffffffu) | (2u << 30);  // addr[56:32] | type=2
  i32x8 g1;
  g1[0] = (int)((1u << 16) | (1u << 20) | (4u << 22) | (3u << 25));
  g1[1] = (int)((tensor_d0 & 0xffffu) << 16);
  g1[2] = (int)(((tensor_d0 >> 16) & 0xffffu) | ((tensor_d1 & 0xffffu) << 16));
  g1[3] = (int)(((tensor_d1 >> 16) & 0xffffu) | (tile_d0 << 16));
  g1[4] = (int)(tile_d1);                                  // tile_dim1 | tile_dim2=0
  g1[5] = (int)(stride0 & 0xffffffffull);
  g1[6] = (int)((stride0 >> 32) & 0xffffull);
  g1[7] = 0;
  i32x4 z4 = {};
  i32x8 z8 = {};
  __builtin_amdgcn_tensor_load_to_lds(g0, g1, z4, z4, z8, 0);
}

// ---------------------------------------------------------------- Wo -> f16
__global__ void __launch_bounds__(256) cvt_wo_kernel(const float* __restrict__ src,
                                                     _Float16* __restrict__ dst, int n) {
  int i = blockIdx.x * 256 + threadIdx.x;
  if (i < n) dst[i] = (_Float16)src[i];
}

// ------------------------------------------- QKV projection: X[65536x64] @ W^T
__global__ void __launch_bounds__(256) qkv_proj_kernel(
    const float* __restrict__ q, const float* __restrict__ k, const float* __restrict__ v,
    const float* __restrict__ Wq, const float* __restrict__ Wk, const float* __restrict__ Wv,
    _Float16* __restrict__ qh, _Float16* __restrict__ kh, _Float16* __restrict__ vt) {
  const int lane = threadIdx.x & 31, wave = threadIdx.x >> 5;
  const int g = lane >> 4, n16 = lane & 15;
  const int tile = blockIdx.x * 8 + wave;  // one 16-row tile per wave
  const int r0 = tile * 16;
  const int bh = r0 / S_;                  // head slices are contiguous 2048-row blocks
  const int s0 = r0 % S_;

  v16h aq0 = load_a_f32(q, D_, r0 + n16, 0, g), aq1 = load_a_f32(q, D_, r0 + n16, 32, g);
  v16h ak0 = load_a_f32(k, D_, r0 + n16, 0, g), ak1 = load_a_f32(k, D_, r0 + n16, 32, g);
  v16h av0 = load_a_f32(v, D_, r0 + n16, 0, g), av1 = load_a_f32(v, D_, r0 + n16, 32, g);

#pragma unroll
  for (int ct = 0; ct < 4; ++ct) {
    const int n = ct * 16 + n16;
    {
      v8f acc = {};
      acc = wmma_f16(aq0, load_bt_f32(Wq + n * D_ + g * 16), acc);
      acc = wmma_f16(aq1, load_bt_f32(Wq + n * D_ + 32 + g * 16), acc);
      store_c_f16(qh + (size_t)(r0 + 8 * g) * D_ + n, D_, acc);
    }
    {
      v8f acc = {};
      acc = wmma_f16(ak0, load_bt_f32(Wk + n * D_ + g * 16), acc);
      acc = wmma_f16(ak1, load_bt_f32(Wk + n * D_ + 32 + g * 16), acc);
      store_c_f16(kh + (size_t)(r0 + 8 * g) * D_ + n, D_, acc);
    }
    {  // V stored transposed [BH][D][S]: per-lane stride-1 -> merges into b128 stores
      v8f acc = {};
      acc = wmma_f16(av0, load_bt_f32(Wv + n * D_ + g * 16), acc);
      acc = wmma_f16(av1, load_bt_f32(Wv + n * D_ + 32 + g * 16), acc);
      store_c_f16(vt + (size_t)bh * D_ * S_ + (size_t)n * S_ + (s0 + 8 * g), 1, acc);
    }
  }
}

// ----------------------------------------------------- flash attention per head
__global__ void __launch_bounds__(256) attn_kernel(
    const _Float16* __restrict__ qh, const _Float16* __restrict__ kh,
    const _Float16* __restrict__ vt, const int* __restrict__ mask,
    _Float16* __restrict__ oh) {
  __shared__ _Float16 kbuf[2][64 * KP_];     // K tile  [key][d],  TDM-padded pitch 72
  __shared__ _Float16 vbuf[2][64 * KP_];     // Vt tile [d][key],  TDM-padded pitch 72
  __shared__ _Float16 ldsP[8][16][68];       // per-wave P transpose, conflict-free pitch
  const int lane = threadIdx.x & 31, wave = threadIdx.x >> 5;
  const int g = lane >> 4, n16 = lane & 15;
  const int bh = blockIdx.x;
  const int b  = bh / H_;
  const int q0 = blockIdx.y * 128 + wave * 16;

  const _Float16* Q  = qh + (size_t)bh * S_ * D_;
  const _Float16* K  = kh + (size_t)bh * S_ * D_;
  const _Float16* Vt = vt + (size_t)bh * (size_t)D_ * S_;
  const int* M = mask + (size_t)b * S_ * S_;

  v16h aq0 = load_a_f16(Q, D_, q0 + n16, 0, g);
  v16h aq1 = load_a_f16(Q, D_, q0 + n16, 32, g);

  float mrow[8], lrow[8];
  v8f accO[4] = {};
#pragma unroll
  for (int r = 0; r < 8; ++r) { mrow[r] = -3.0e38f; lrow[r] = 0.0f; }

  constexpr int NT = S_ / 64;
  // One DMA issuer per block; all waves consume from LDS after the barrier.
  if (wave == 0) {
    tdm_load_2d(K, (unsigned)(uintptr_t)&kbuf[0][0], D_, S_, D_, 64, 64);
    tdm_load_2d(Vt, (unsigned)(uintptr_t)&vbuf[0][0], S_, D_, S_, 64, 64);
  }

  for (int kt = 0; kt < NT; ++kt) {
    const int cur = kt & 1;
    const int kbase = kt * 64;
    if (wave == 0) __builtin_amdgcn_s_wait_tensorcnt(0);
    __syncthreads();                       // tile kt resident; tile kt-1 reads all done
    if (wave == 0 && kt + 1 < NT) {        // overlap next tile's DMA with this compute
      tdm_load_2d(K + (size_t)(kbase + 64) * D_,
                  (unsigned)(uintptr_t)&kbuf[cur ^ 1][0], D_, S_, D_, 64, 64);
      tdm_load_2d(Vt + (size_t)(kbase + 64),
                  (unsigned)(uintptr_t)&vbuf[cur ^ 1][0], S_, D_, S_, 64, 64);
    }
    __builtin_prefetch(M + (size_t)(q0 + n16) * S_ + kbase + g * 32, 0, 3);

    // S = Q K^T for this 16x64 stripe (4 column tiles, 2 k-steps) from LDS
    v8f sc[4];
#pragma unroll
    for (int nt = 0; nt < 4; ++nt) {
      const int kl = nt * 16 + n16;        // key within tile
      v8f s = {};
      s = wmma_f16(aq0, load_bt_f16(&kbuf[cur][kl * KP_ + g * 16]), s);
      s = wmma_f16(aq1, load_bt_f16(&kbuf[cur][kl * KP_ + 32 + g * 16]), s);
#pragma unroll
      for (int r = 0; r < 8; ++r) {        // mask + scale (faithful to reference)
        const int qg = q0 + r + 8 * g;
        const int mv = M[(size_t)qg * S_ + kbase + kl];
        s[r] = ((mv == 0) ? -1.0e20f : s[r]) * SCALE_;
      }
      sc[nt] = s;
    }

    // online softmax: per-lane stats cover rows r+8g; reduce across N via shfl_xor<16
#pragma unroll
    for (int r = 0; r < 8; ++r) {
      float mx = fmaxf(fmaxf(sc[0][r], sc[1][r]), fmaxf(sc[2][r], sc[3][r]));
      mx = fmaxf(mx, __shfl_xor(mx, 1));
      mx = fmaxf(mx, __shfl_xor(mx, 2));
      mx = fmaxf(mx, __shfl_xor(mx, 4));
      mx = fmaxf(mx, __shfl_xor(mx, 8));
      const float mnew = fmaxf(mrow[r], mx);
      const float corr = __expf(mrow[r] - mnew);
      mrow[r] = mnew;
      float rs = 0.0f;
#pragma unroll
      for (int nt = 0; nt < 4; ++nt) {
        const float p = __expf(sc[nt][r] - mnew);
        sc[nt][r] = p;
        rs += p;
      }
      rs += __shfl_xor(rs, 1);
      rs += __shfl_xor(rs, 2);
      rs += __shfl_xor(rs, 4);
      rs += __shfl_xor(rs, 8);
      lrow[r] = lrow[r] * corr + rs;
#pragma unroll
      for (int ct = 0; ct < 4; ++ct) accO[ct][r] *= corr;
    }

    // C-layout P -> per-wave LDS tile -> A-layout fragments
#pragma unroll
    for (int nt = 0; nt < 4; ++nt)
#pragma unroll
      for (int r = 0; r < 8; ++r)
        ldsP[wave][r + 8 * g][nt * 16 + n16] = (_Float16)sc[nt][r];

#pragma unroll
    for (int ks = 0; ks < 2; ++ks) {
      v16h ap;
      const _Float16* lp = &ldsP[wave][n16][ks * 32 + g * 16];
#pragma unroll
      for (int i = 0; i < 16; ++i) ap[i] = lp[i];
#pragma unroll
      for (int ct = 0; ct < 4; ++ct) {
        const int d = ct * 16 + n16;
        v16h bv = load_bt_f16(&vbuf[cur][d * KP_ + ks * 32 + g * 16]);
        accO[ct] = wmma_f16(ap, bv, accO[ct]);
      }
    }
  }

  _Float16* O = oh + (size_t)bh * S_ * D_;
#pragma unroll
  for (int ct = 0; ct < 4; ++ct) {
    v8f on;
#pragma unroll
    for (int r = 0; r < 8; ++r) on[r] = accO[ct][r] / lrow[r];
    store_c_f16(O + (size_t)(q0 + 8 * g) * D_ + ct * 16 + n16, D_, on);
  }
}

// ------------------------------------- output projection: [4096x1024] @ Wo^T + bo
__global__ void __launch_bounds__(256) out_proj_kernel(
    const _Float16* __restrict__ oh, const _Float16* __restrict__ wo16,
    const float* __restrict__ bo, float* __restrict__ out) {
  const int lane = threadIdx.x & 31, wave = threadIdx.x >> 5;
  const int g = lane >> 4, n16 = lane & 15;
  const int job = blockIdx.x * 8 + wave;   // 256 row-tiles x 16 col-groups
  const int r0 = (job >> 4) * 16;
  const int c0 = (job & 15) * 64;

  v8f acc[4] = {};
  for (int ks = 0; ks < E_ / 32; ++ks) {
    v16h a = load_a_f16(oh, E_, r0 + n16, ks * 32, g);
#pragma unroll
    for (int ct = 0; ct < 4; ++ct) {
      const int n = c0 + ct * 16 + n16;
      acc[ct] = wmma_f16(a, load_bt_f16(wo16 + (size_t)n * E_ + ks * 32 + g * 16), acc[ct]);
    }
  }
#pragma unroll
  for (int ct = 0; ct < 4; ++ct) {
    const int col = c0 + ct * 16 + n16;
    const float bias = bo[col];
    float o[8];
#pragma unroll
    for (int r = 0; r < 8; ++r) o[r] = acc[ct][r] + bias;
#pragma unroll
    for (int r = 0; r < 8; ++r) out[(size_t)(r0 + r + 8 * g) * E_ + col] = o[r];
  }
}

extern "C" void kernel_launch(void* const* d_in, const int* in_sizes, int n_in,
                              void* d_out, int out_size, void* d_ws, size_t ws_size,
                              hipStream_t stream) {
  const float* q    = (const float*)d_in[0];
  const float* k    = (const float*)d_in[1];
  const float* v    = (const float*)d_in[2];
  const int*   mask = (const int*)d_in[3];
  const float* Wq   = (const float*)d_in[4];
  const float* Wk   = (const float*)d_in[5];
  const float* Wv   = (const float*)d_in[6];
  const float* Wo   = (const float*)d_in[7];
  const float* bo   = (const float*)d_in[8];
  float* out = (float*)d_out;

  _Float16* ws   = (_Float16*)d_ws;
  _Float16* qh   = ws + WS_Q;
  _Float16* kh   = ws + WS_K;
  _Float16* vt   = ws + WS_VT;
  _Float16* oh   = ws + WS_O;
  _Float16* wo16 = ws + WS_WO;

  cvt_wo_kernel<<<(E_ * E_ + 255) / 256, 256, 0, stream>>>(Wo, wo16, E_ * E_);
  qkv_proj_kernel<<<ROWS_ / 16 / 8, 256, 0, stream>>>(q, k, v, Wq, Wk, Wv, qh, kh, vt);
  attn_kernel<<<dim3(BH_, S_ / 128), 256, 0, stream>>>(qh, kh, vt, mask, oh);
  out_proj_kernel<<<(256 * 16) / 8, 256, 0, stream>>>(oh, wo16, bo, out);
}
